// LSTM_11759620456659
// MI455X (gfx1250) — matrix operand order repaired and checked by reading, not compiled, not run on hardware
//
#include <hip/hip_runtime.h>
#include <math.h>

// ---------------- types for WMMA ----------------
typedef __bf16 bf16_t;
typedef bf16_t v16bf __attribute__((ext_vector_type(16)));
typedef float  v8f   __attribute__((ext_vector_type(8)));

#define B_ 256
#define D_ 512
#define L_ 196
#define N_ 1024
#define M_ 512
#define K_ 10000

__device__ __forceinline__ float sigf(float x) { return 1.0f / (1.0f + __expf(-x)); }

// ============================================================
// Kernel 1: fused attention: e, softmax(alpha), beta, z
// one block per batch element, 256 threads (8 waves of 32)
// ============================================================
__global__ __launch_bounds__(256) void attn_fused(
    const float* __restrict__ a_i, const float* __restrict__ h,
    const float* __restrict__ f_att_w, const float* __restrict__ f_att_b,
    const float* __restrict__ gate_w, const float* __restrict__ gate_b,
    float* __restrict__ alpha_out, float* __restrict__ z_out)
{
    __shared__ float red[256];
    __shared__ float red2[256];
    __shared__ float s_alpha[L_];
    __shared__ float s_hw;
    __shared__ float s_beta;

    const int b = blockIdx.x;
    const int t = threadIdx.x;
    const float* hb = h + (size_t)b * N_;

    // ---- reductions: h . w_h  and  h . gate_w ----
    float s1 = 0.f, s2 = 0.f;
    for (int j = t; j < N_; j += 256) {
        float hv = hb[j];
        s1 += hv * f_att_w[D_ + j];
        s2 += hv * gate_w[j];
    }
    red[t] = s1; red2[t] = s2;
    __syncthreads();
    for (int off = 128; off; off >>= 1) {
        if (t < off) { red[t] += red[t + off]; red2[t] += red2[t + off]; }
        __syncthreads();
    }
    if (t == 0) {
        s_hw   = red[0] + f_att_b[0];
        s_beta = sigf(red2[0] + gate_b[0]);
    }
    __syncthreads();

    // ---- e[l] = sum_d a_i[b,d,l] * w_a[d] + hw   (coalesced over l) ----
    float e = -3.0e38f;
    if (t < L_) {
        float acc = 0.f;
        const float* ab = a_i + (size_t)b * D_ * L_ + t;
        for (int d = 0; d < D_; ++d) acc += ab[(size_t)d * L_] * f_att_w[d];
        e = acc + s_hw;
    }

    // ---- softmax over L ----
    red[t] = e;
    __syncthreads();
    for (int off = 128; off; off >>= 1) {
        if (t < off) red[t] = fmaxf(red[t], red[t + off]);
        __syncthreads();
    }
    float mx = red[0];
    __syncthreads();
    float ex = (t < L_) ? __expf(e - mx) : 0.f;
    red[t] = ex;
    __syncthreads();
    for (int off = 128; off; off >>= 1) {
        if (t < off) red[t] += red[t + off];
        __syncthreads();
    }
    float inv = 1.f / red[0];
    if (t < L_) {
        float al = ex * inv;
        s_alpha[t] = al;
        alpha_out[(size_t)b * L_ + t] = al;
    }
    __syncthreads();

    // ---- z[b,d] = beta * sum_l alpha[l] * a_i[b,d,l] : wave-per-d ----
    const int lane = t & 31;
    const int wv   = t >> 5;
    const float beta = s_beta;
    for (int d = wv; d < D_; d += 8) {
        const float* ab = a_i + ((size_t)b * D_ + d) * L_;
        float acc = 0.f;
        for (int lb = 0; lb < L_; lb += 32) {
            int l = lb + lane;
            if (l < L_) acc += s_alpha[l] * ab[l];
        }
        for (int off = 16; off; off >>= 1) acc += __shfl_xor(acc, off, 32);
        if (lane == 0) z_out[(size_t)b * D_ + d] = beta * acc;
    }
}

// ============================================================
// WMMA fragment helpers (ISA 16-bit operand VGPR layouts)
// ============================================================
__device__ __forceinline__ v16bf pack16(float4 x0, float4 x1, float4 x2, float4 x3)
{
    v16bf v;
    v[0]  = (bf16_t)x0.x; v[1]  = (bf16_t)x0.y; v[2]  = (bf16_t)x0.z; v[3]  = (bf16_t)x0.w;
    v[4]  = (bf16_t)x1.x; v[5]  = (bf16_t)x1.y; v[6]  = (bf16_t)x1.z; v[7]  = (bf16_t)x1.w;
    v[8]  = (bf16_t)x2.x; v[9]  = (bf16_t)x2.y; v[10] = (bf16_t)x2.z; v[11] = (bf16_t)x2.w;
    v[12] = (bf16_t)x3.x; v[13] = (bf16_t)x3.y; v[14] = (bf16_t)x3.z; v[15] = (bf16_t)x3.w;
    return v;
}

// A-matrix 16x32: lane(lh,kh) holds A[m, kh*8 .. +8) and A[m, 16+kh*8 .. +8)
__device__ __forceinline__ v16bf load_a_frag(const float* __restrict__ p, int kh)
{
    const float4* q0 = (const float4*)(p + kh * 8);
    const float4* q1 = (const float4*)(p + 16 + kh * 8);
    return pack16(q0[0], q0[1], q1[0], q1[1]);
}

// B-matrix 32x16: lane(lh,kh) holds W[n, kh*8 .. +8) and W[n, 16+kh*8 .. +8)  (B[k][n] = W[n][k])
__device__ __forceinline__ v16bf load_b_frag(const float* __restrict__ p, int kh)
{
    const float4* q0 = (const float4*)(p + kh * 8);
    const float4* q1 = (const float4*)(p + 16 + kh * 8);
    return pack16(q0[0], q0[1], q1[0], q1[1]);
}

static const float4 f4zero = {0.f, 0.f, 0.f, 0.f};

// ============================================================
// Kernel 2: WMMA GEMM:  C[M,N] (+)= A[M,K] . W[N,K]^T + bias
// block = 8 waves -> 64(M) x 64(N); each wave: 32x16 tile,
// two A fragments sharing one B (weight) fragment.
// Requires M % 64 == 0 and K % 16 == 0 (true for all calls).
// ============================================================
__global__ __launch_bounds__(256) void gemm_bf16_wmma(
    float* __restrict__ C, const float* __restrict__ A,
    const float* __restrict__ W, const float* __restrict__ bias,
    int M, int N, int K, int accum)
{
    const int lane = threadIdx.x & 31;
    const int wv   = threadIdx.x >> 5;
    const int lh   = lane & 15;   // row/col within fragment
    const int kh   = lane >> 4;   // k-half selector

    const int mBase = blockIdx.y * 64 + (wv & 1) * 32;
    const int nBase = blockIdx.x * 64 + (wv >> 1) * 16;
    const int n  = nBase + lh;
    const int nc = (n < N) ? n : (N - 1);   // clamp for safe loads; store guarded

    const float* pA0 = A + (size_t)(mBase + lh) * K;
    const float* pA1 = A + (size_t)(mBase + 16 + lh) * K;
    const float* pW  = W + (size_t)nc * K;

    v8f acc0 = {0.f, 0.f, 0.f, 0.f, 0.f, 0.f, 0.f, 0.f};
    v8f acc1 = {0.f, 0.f, 0.f, 0.f, 0.f, 0.f, 0.f, 0.f};

    const int Kfast = K & ~31;
    int kk = 0;
#pragma unroll 2
    for (; kk < Kfast; kk += 32) {
        v16bf af0 = load_a_frag(pA0 + kk, kh);
        v16bf af1 = load_a_frag(pA1 + kk, kh);
        v16bf bf  = load_b_frag(pW + kk, kh);
        acc0 = __builtin_amdgcn_wmma_f32_16x16x32_bf16(
            false, af0, false, bf, (short)0, acc0, false, false);
        acc1 = __builtin_amdgcn_wmma_f32_16x16x32_bf16(
            false, af1, false, bf, (short)0, acc1, false, false);
    }

    if (kk < K) {
        // K tail (K % 16 == 0): each 8-float A chunk / 8-float W chunk is
        // uniformly valid or invalid -> chunk-granular vector loads.
        const int ka = kk + kh * 8;        // A chunk 0
        const int kb = kk + 16 + kh * 8;   // A chunk 1
        const int kw = kk + kh * 16;       // unused; W chunks use ka / kb
        float4 a00 = f4zero, a01 = f4zero, a10 = f4zero, a11 = f4zero;
        float4 b00 = f4zero, b01 = f4zero, b10 = f4zero, b11 = f4zero;
        float4 c00 = f4zero, c01 = f4zero;
        if (ka + 8 <= K) {
            const float4* q0 = (const float4*)(pA0 + ka);
            const float4* q1 = (const float4*)(pA1 + ka);
            a00 = q0[0]; a01 = q0[1];
            b00 = q1[0]; b01 = q1[1];
        }
        if (kb + 8 <= K) {
            const float4* q0 = (const float4*)(pA0 + kb);
            const float4* q1 = (const float4*)(pA1 + kb);
            a10 = q0[0]; a11 = q0[1];
            b10 = q1[0]; b11 = q1[1];
        }
        v16bf af0 = pack16(a00, a01, a10, a11);
        v16bf af1 = pack16(b00, b01, b10, b11);
        float4 w00 = c00, w01 = c01, w10 = c00, w11 = c01;
        if (ka + 8 <= K) {
            const float4* q = (const float4*)(pW + ka);
            w00 = q[0]; w01 = q[1];
        }
        if (kb + 8 <= K) {
            const float4* q = (const float4*)(pW + kb);
            w10 = q[0]; w11 = q[1];
        }
        v16bf bf = pack16(w00, w01, w10, w11);
        acc0 = __builtin_amdgcn_wmma_f32_16x16x32_bf16(
            false, af0, false, bf, (short)0, acc0, false, false);
        acc1 = __builtin_amdgcn_wmma_f32_16x16x32_bf16(
            false, af1, false, bf, (short)0, acc1, false, false);
    }

    // C/D layout: lane 0-15 -> M=r, lane 16-31 -> M=8+r ; N = nBase+lh
    if (n < N) {
        float bi = bias ? bias[n] : 0.f;
#pragma unroll
        for (int r = 0; r < 8; ++r) {
            int row0 = mBase + kh * 8 + r;
            int row1 = row0 + 16;
            size_t off0 = (size_t)row0 * N + n;
            size_t off1 = (size_t)row1 * N + n;
            float v0 = acc0[r] + bi;
            float v1 = acc1[r] + bi;
            if (accum) { v0 += C[off0]; v1 += C[off1]; }
            C[off0] = v0;
            C[off1] = v1;
        }
    }
}

// ============================================================
// Kernel 3: x = concat([w_emb, h, z], dim=1)  -> (B, 2048)
// ============================================================
__global__ __launch_bounds__(256) void concat_x(
    const float* __restrict__ w_emb, const float* __restrict__ h,
    const float* __restrict__ z, float* __restrict__ x)
{
    int idx = blockIdx.x * 256 + threadIdx.x;           // B_*2048
    int b = idx >> 11, col = idx & 2047;
    float v;
    if (col < 512)       v = w_emb[(size_t)b * 512 + col];
    else if (col < 1536) v = h[(size_t)b * 1024 + (col - 512)];
    else                 v = z[(size_t)b * 512 + (col - 1536)];
    x[idx] = v;
}

// ============================================================
// Kernel 4: LSTM cell elementwise
// ============================================================
__global__ __launch_bounds__(256) void lstm_cell(
    const float* __restrict__ gates, const float* __restrict__ c_prev,
    float* __restrict__ hn, float* __restrict__ cn)
{
    int idx = blockIdx.x * 256 + threadIdx.x;           // B_*N_
    int b = idx >> 10, n = idx & 1023;
    const float* g = gates + (size_t)b * 4 * N_;
    float ig = sigf(g[n]);
    float fg = sigf(g[N_ + n]);
    float gg = tanhf(g[2 * N_ + n]);
    float og = sigf(g[3 * N_ + n]);
    float c  = fg * c_prev[idx] + ig * gg;
    float hh = og * tanhf(c);
    cn[idx] = c;
    hn[idx] = hh;
}

// ============================================================
// Kernel 5: deep_out init = w_emb + Lh_b + Lz_b
// ============================================================
__global__ __launch_bounds__(256) void deep_init(
    const float* __restrict__ w_emb, const float* __restrict__ Lh_b,
    const float* __restrict__ Lz_b, float* __restrict__ deep)
{
    int idx = blockIdx.x * 256 + threadIdx.x;           // B_*M_
    int col = idx & 511;
    deep[idx] = w_emb[idx] + Lh_b[col] + Lz_b[col];
}

// ============================================================
// launcher
// ============================================================
extern "C" void kernel_launch(void* const* d_in, const int* in_sizes, int n_in,
                              void* d_out, int out_size, void* d_ws, size_t ws_size,
                              hipStream_t stream)
{
    const float* a_i     = (const float*)d_in[0];
    const float* inp     = (const float*)d_in[1];
    const float* h_prev  = (const float*)d_in[2];
    const float* c_prev  = (const float*)d_in[3];
    const float* f_att_w = (const float*)d_in[4];
    const float* f_att_b = (const float*)d_in[5];
    const float* gate_w  = (const float*)d_in[6];
    const float* gate_b  = (const float*)d_in[7];
    const float* E_w     = (const float*)d_in[8];
    const float* E_b     = (const float*)d_in[9];
    const float* W_ih    = (const float*)d_in[10];
    const float* W_hh    = (const float*)d_in[11];
    const float* b_ih    = (const float*)d_in[12];
    const float* b_hh    = (const float*)d_in[13];
    const float* Lh_w    = (const float*)d_in[14];
    const float* Lh_b    = (const float*)d_in[15];
    const float* Lz_w    = (const float*)d_in[16];
    const float* Lz_b    = (const float*)d_in[17];
    const float* Lo_w    = (const float*)d_in[18];
    const float* Lo_b    = (const float*)d_in[19];

    float* out   = (float*)d_out;
    float* p_yt  = out;                              // 256*10000
    float* hn    = out + (size_t)B_ * K_;            // 256*1024
    float* cn    = hn + (size_t)B_ * N_;             // 256*1024
    float* alpha = cn + (size_t)B_ * N_;             // 256*196

    float* ws    = (float*)d_ws;
    float* z     = ws;                               // 256*512
    float* w_emb = z + (size_t)B_ * D_;              // 256*512
    float* x     = w_emb + (size_t)B_ * M_;          // 256*2048
    float* gates = x + (size_t)B_ * 2048;            // 256*4096
    float* deep  = gates + (size_t)B_ * 4 * N_;      // 256*512

    // 1) attention: alpha, z
    attn_fused<<<B_, 256, 0, stream>>>(a_i, h_prev, f_att_w, f_att_b,
                                       gate_w, gate_b, alpha, z);

    // 2) w_emb = inp @ E_w^T + E_b              (256x512, K=10000)
    gemm_bf16_wmma<<<dim3(M_ / 64, B_ / 64), 256, 0, stream>>>(
        w_emb, inp, E_w, E_b, B_, M_, K_, 0);

    // 3) x = [w_emb, h, z]
    concat_x<<<(B_ * 2048) / 256, 256, 0, stream>>>(w_emb, h_prev, z, x);

    // 4) gates = x @ W_ih^T + b_ih  + h @ W_hh^T + b_hh   (256x4096)
    gemm_bf16_wmma<<<dim3(4 * N_ / 64, B_ / 64), 256, 0, stream>>>(
        gates, x, W_ih, b_ih, B_, 4 * N_, 2048, 0);
    gemm_bf16_wmma<<<dim3(4 * N_ / 64, B_ / 64), 256, 0, stream>>>(
        gates, h_prev, W_hh, b_hh, B_, 4 * N_, N_, 1);

    // 5) LSTM cell -> hn, cn
    lstm_cell<<<(B_ * N_) / 256, 256, 0, stream>>>(gates, c_prev, hn, cn);

    // 6) deep_out = w_emb + Lh_b + Lz_b + hn @ Lh_w^T + z @ Lz_w^T
    deep_init<<<(B_ * M_) / 256, 256, 0, stream>>>(w_emb, Lh_b, Lz_b, deep);
    gemm_bf16_wmma<<<dim3(M_ / 64, B_ / 64), 256, 0, stream>>>(
        deep, hn, Lh_w, nullptr, B_, M_, N_, 1);
    gemm_bf16_wmma<<<dim3(M_ / 64, B_ / 64), 256, 0, stream>>>(
        deep, z, Lz_w, nullptr, B_, M_, D_, 1);

    // 7) p_yt = deep_out @ Lo_w^T + Lo_b        (256x10000, K=512)
    gemm_bf16_wmma<<<dim3((K_ + 63) / 64, B_ / 64), 256, 0, stream>>>(
        p_yt, deep, Lo_w, Lo_b, B_, K_, M_, 0);
}
